// TemporalGraphSAINT_60507499266856
// MI455X (gfx1250) — matrix operand order repaired and checked by reading, not compiled, not run on hardware
//
#include <hip/hip_runtime.h>

typedef __attribute__((ext_vector_type(16))) _Float16 v16h;
typedef __attribute__((ext_vector_type(8)))  _Float16 v8h;
typedef __attribute__((ext_vector_type(4)))  _Float16 v4h;
typedef __attribute__((ext_vector_type(8)))  float    v8f;
typedef __attribute__((ext_vector_type(4)))  float    f32x4;

#define IN_CH   128
#define HID_CH  256
#define OUT_CH  128
#define S_STEP  500000
#define ROWS_PER_BLOCK 128            // 8 waves * 16 rows
#define THREADS 256
#define NBLOCKS ((S_STEP + ROWS_PER_BLOCK - 1) / ROWS_PER_BLOCK)   // 3907
#define S_PAD   (NBLOCKS * ROWS_PER_BLOCK)                         // 500096

// d_ws layout:
//   [W1 f16: 65536 B][W2 f16: 65536 B][H f16: S_PAD*256*2 B][dump: 4 KB]
#define W1H_OFF 0
#define W2H_OFF 65536
#define H_OFF   131072
#define DUMP_OFF (H_OFF + (size_t)S_PAD * HID_CH * 2)
// ws_size needed: DUMP_OFF + 4096  (~256.2 MB)

// ---------------------------------------------------------------------------
// One-shot weight conversion f32 -> f16 (runs once per launch, ~256 KB read).
// ---------------------------------------------------------------------------
__global__ __launch_bounds__(THREADS) void tgs_prep(
    const float* __restrict__ W1, const float* __restrict__ W2,
    _Float16* __restrict__ W1h, _Float16* __restrict__ W2h)
{
  const int idx    = blockIdx.x * THREADS + threadIdx.x;
  const int stride = gridDim.x * THREADS;
  for (int i = idx; i < (HID_CH * IN_CH) / 4; i += stride) {
    f32x4 v = ((const f32x4*)W1)[i];
    v4h h;
    h[0] = (_Float16)v[0]; h[1] = (_Float16)v[1];
    h[2] = (_Float16)v[2]; h[3] = (_Float16)v[3];
    ((v4h*)W1h)[i] = h;
  }
  for (int i = idx; i < (OUT_CH * HID_CH) / 4; i += stride) {
    f32x4 v = ((const f32x4*)W2)[i];
    v4h h;
    h[0] = (_Float16)v[0]; h[1] = (_Float16)v[1];
    h[2] = (_Float16)v[2]; h[3] = (_Float16)v[3];
    ((v4h*)W2h)[i] = h;
  }
}

// ---------------------------------------------------------------------------
// Layer 1: H[lr, :] = relu(x[n_id[lr], :] @ W1^T + b1), H stored as f16.
// H is padded to S_PAD rows so every store is unconditional (no tail masks).
// Wave owns 16 batch rows; A fragments live in VGPRs across all N-tiles.
// W1h staged to LDS with a 16-half-block XOR swizzle.
// ---------------------------------------------------------------------------
__global__ __launch_bounds__(THREADS) void tgs_layer1(
    const float*    __restrict__ xcur,
    const int*      __restrict__ nid,
    const _Float16* __restrict__ W1h,
    const float*    __restrict__ b1,
    _Float16*       __restrict__ H)
{
  __shared__ __align__(32) _Float16 w1s[HID_CH * IN_CH];   // 64 KB
  __shared__ float b1s[HID_CH];                            // 1 KB

  const int tid = threadIdx.x;
  {
    const v8h* src = (const v8h*)W1h;
    #pragma unroll
    for (int t = 0; t < (HID_CH * IN_CH) / (8 * THREADS); ++t) {  // 16 iters
      int i = t * THREADS + tid;
      v8h v = src[i];
      int e  = 8 * i;
      int n  = e >> 7;
      int k  = e & 127;
      int kb = k >> 4;
      int dst = n * IN_CH + (((kb ^ (n & 7)) << 4) | (k & 15));
      *(v8h*)&w1s[dst] = v;
    }
    b1s[tid] = b1[tid];   // HID_CH == THREADS
  }
  __syncthreads();

  const int lane   = tid & 31;
  const int wave   = tid >> 5;
  const int mlane  = lane & 15;
  const int hiHalf = lane >> 4;
  const long rowbase = (long)blockIdx.x * ROWS_PER_BLOCK + (long)wave * 16;

  long arow = rowbase + mlane;
  if (arow > S_STEP - 1) arow = S_STEP - 1;       // clamp gather index only
  const float* xrow = xcur + (long)nid[arow] * IN_CH;

  // A fragments: 4 k-tiles of 32. Contiguous order: lanes 0-15 hold
  // K in {0..15}+kbase, lanes 16-31 hold {16..31}+kbase.
  v16h a[4];
  const int koffLo = hiHalf ? 16 : 0;
  #pragma unroll
  for (int kt = 0; kt < 4; ++kt) {
    const f32x4* p = (const f32x4*)(xrow + kt * 32 + koffLo);
    f32x4 f0 = p[0], f1 = p[1], f2 = p[2], f3 = p[3];
    #pragma unroll
    for (int j = 0; j < 4; ++j) {
      a[kt][j]      = (_Float16)f0[j];
      a[kt][4 + j]  = (_Float16)f1[j];
      a[kt][8 + j]  = (_Float16)f2[j];
      a[kt][12 + j] = (_Float16)f3[j];
    }
  }

  _Float16* hbase = H + rowbase * HID_CH + mlane + (long)hiHalf * 8 * HID_CH;

  #pragma unroll 1
  for (int nt = 0; nt < HID_CH / 16; ++nt) {
    const int ncol = nt * 16 + mlane;
    const float bias = b1s[ncol];

    // Fetch all 4 B fragments first -> one DS wait, then 4 chained WMMAs.
    v16h bf[4];
    #pragma unroll
    for (int kt = 0; kt < 4; ++kt) {
      const int kb = kt * 2 + hiHalf;
      bf[kt] = *(const v16h*)&w1s[ncol * IN_CH + ((kb ^ (ncol & 7)) << 4)];
    }
    v8f c;
    #pragma unroll
    for (int i = 0; i < 8; ++i) c[i] = bias;
    #pragma unroll
    for (int kt = 0; kt < 4; ++kt)
      c = __builtin_amdgcn_wmma_f32_16x16x32_f16(
              false, a[kt], false, bf[kt], (short)0, c, false, false);

    // C layout: VGPR i -> row (rowbase + i + 8*hiHalf), col = ncol.
    // Rows < S_PAD by construction: unconditional stores.
    _Float16* hp = hbase + nt * 16;
    #pragma unroll
    for (int i = 0; i < 8; ++i)
      hp[(long)i * HID_CH] = (_Float16)fmaxf(c[i], 0.0f);
  }
}

// ---------------------------------------------------------------------------
// Layer 2: x_out[n_id[lr], :] = H[lr, :] @ W2^T + b2 (scatter, last write
// wins). Tail rows scatter into a dump area via per-lane address select --
// no masked stores, no exec manipulation.
// ---------------------------------------------------------------------------
__global__ __launch_bounds__(THREADS) void tgs_layer2(
    const _Float16* __restrict__ H,
    const int*      __restrict__ nid,
    const _Float16* __restrict__ W2h,
    const float*    __restrict__ b2,
    float*          __restrict__ xout,
    float*          __restrict__ dump)
{
  __shared__ __align__(32) _Float16 w2s[OUT_CH * HID_CH];  // 64 KB
  __shared__ float b2s[OUT_CH];                            // 512 B

  const int tid = threadIdx.x;
  {
    const v8h* src = (const v8h*)W2h;
    #pragma unroll
    for (int t = 0; t < (OUT_CH * HID_CH) / (8 * THREADS); ++t) {  // 16 iters
      int i = t * THREADS + tid;
      v8h v = src[i];
      int e  = 8 * i;
      int n  = e >> 8;
      int k  = e & 255;
      int kb = k >> 4;
      int dst = n * HID_CH + (((kb ^ (n & 15)) << 4) | (k & 15));
      *(v8h*)&w2s[dst] = v;
    }
    if (tid < OUT_CH) b2s[tid] = b2[tid];
  }
  __syncthreads();

  const int lane   = tid & 31;
  const int wave   = tid >> 5;
  const int mlane  = lane & 15;
  const int hiHalf = lane >> 4;
  const long rowbase = (long)blockIdx.x * ROWS_PER_BLOCK + (long)wave * 16;

  // H is padded: rows rowbase..rowbase+15 always valid, no clamp needed.
  const _Float16* hrow = H + (rowbase + mlane) * HID_CH;

  // A fragments: 8 k-tiles of 32, two 16B half8 chunks each.
  v16h a[8];
  const int koffLo = hiHalf ? 16 : 0;
  #pragma unroll
  for (int kt = 0; kt < 8; ++kt) {
    const v8h* p = (const v8h*)(hrow + kt * 32 + koffLo);
    v8h lo = p[0];      // K = kbase+koffLo .. +7
    v8h hi = p[1];      // K = kbase+8+koffLo .. +7
    #pragma unroll
    for (int j = 0; j < 8; ++j) { a[kt][j] = lo[j]; a[kt][8 + j] = hi[j]; }
  }

  // Per-lane scatter base pointers (nt-invariant). Out-of-range rows are
  // redirected to the dump area: address select, not a masked store.
  float* bp[8];
  #pragma unroll
  for (int i = 0; i < 8; ++i) {
    long lr  = rowbase + i + hiHalf * 8;
    long lrc = (lr < S_STEP) ? lr : (S_STEP - 1);   // safe nid read
    int  gid = nid[lrc];
    bp[i] = (lr < S_STEP) ? (xout + (long)gid * OUT_CH) : dump;
  }

  #pragma unroll 1
  for (int nt = 0; nt < OUT_CH / 16; ++nt) {
    const int ncol = nt * 16 + mlane;
    const float bias = b2s[ncol];

    v8f c;
    #pragma unroll
    for (int i = 0; i < 8; ++i) c[i] = bias;
    // Two groups of 4: fetch 4 B fragments, then 4 chained WMMAs.
    #pragma unroll
    for (int g = 0; g < 2; ++g) {
      v16h bf[4];
      #pragma unroll
      for (int q = 0; q < 4; ++q) {
        const int kt = g * 4 + q;
        const int kb = kt * 2 + hiHalf;
        bf[q] = *(const v16h*)&w2s[ncol * HID_CH + ((kb ^ (ncol & 15)) << 4)];
      }
      #pragma unroll
      for (int q = 0; q < 4; ++q)
        c = __builtin_amdgcn_wmma_f32_16x16x32_f16(
                false, a[g * 4 + q], false, bf[q], (short)0, c, false, false);
    }
    #pragma unroll
    for (int i = 0; i < 8; ++i)
      bp[i][ncol] = c[i];
  }
}

// ---------------------------------------------------------------------------
// inputs: 0:x [2M*128 f32] 1:edge_index 2:time_index 3:n_id [2*500000 i32]
//         4:W1 [256*128 f32] 5:b1 [256] 6:W2 [128*256 f32] 7:b2 [128]
// d_out:  updated x [2M*128 f32].
// ---------------------------------------------------------------------------
extern "C" void kernel_launch(void* const* d_in, const int* in_sizes, int n_in,
                              void* d_out, int out_size, void* d_ws, size_t ws_size,
                              hipStream_t stream) {
  (void)in_sizes; (void)n_in; (void)ws_size;
  const float* x       = (const float*)d_in[0];
  const int*   nid_all = (const int*)  d_in[3];
  const float* W1      = (const float*)d_in[4];
  const float* b1      = (const float*)d_in[5];
  const float* W2      = (const float*)d_in[6];
  const float* b2      = (const float*)d_in[7];
  float*    xo   = (float*)d_out;
  _Float16* W1h  = (_Float16*)((char*)d_ws + W1H_OFF);
  _Float16* W2h  = (_Float16*)((char*)d_ws + W2H_OFF);
  _Float16* H    = (_Float16*)((char*)d_ws + H_OFF);
  float*    dump = (float*)((char*)d_ws + DUMP_OFF);

  // x_out starts as a copy of x; steps update it in place (stream-serialized).
  (void)hipMemcpyAsync(d_out, (const void*)x, (size_t)out_size * sizeof(float),
                       hipMemcpyDeviceToDevice, stream);
  tgs_prep<<<64, THREADS, 0, stream>>>(W1, W2, W1h, W2h);

  for (int step = 0; step < 2; ++step) {
    const int* nid = nid_all + (size_t)step * S_STEP;
    tgs_layer1<<<NBLOCKS, THREADS, 0, stream>>>(xo, nid, W1h, b1, H);
    tgs_layer2<<<NBLOCKS, THREADS, 0, stream>>>(H, nid, W2h, b2, xo, dump);
  }
}
